// MultiHeadAttention_35356170780991
// MI455X (gfx1250) — matrix operand order repaired and checked by reading, not compiled, not run on hardware
//
#include <hip/hip_runtime.h>

// ---------------------------------------------------------------------------
// MultiHeadAttention for MI455X (gfx1250, wave32, WMMA 16x16x32 f16 + TDM)
// B=2, S=2048, D=1024, H=16, DK=64.
// ---------------------------------------------------------------------------

using half_t = _Float16;
typedef __attribute__((ext_vector_type(16))) _Float16 v16h;
typedef __attribute__((ext_vector_type(8)))  _Float16 v8h;
typedef __attribute__((ext_vector_type(8)))  float    v8f;
typedef __attribute__((ext_vector_type(4)))  unsigned int u32x4;
typedef __attribute__((ext_vector_type(8)))  int i32x8;
typedef __attribute__((ext_vector_type(4)))  int i32x4;

constexpr int B_  = 2;
constexpr int S_  = 2048;
constexpr int D_  = 1024;
constexpr int H_  = 16;
constexpr int DK_ = 64;
constexpr int M_  = B_ * S_;          // 4096 total rows

// ---- workspace layout (in half_t elements) --------------------------------
constexpr size_t WQT_OFF  = 0;                                  // [H,DK,D]
constexpr size_t WKT_OFF  = WQT_OFF + (size_t)H_ * DK_ * D_;
constexpr size_t WVT_OFF  = WKT_OFF + (size_t)H_ * DK_ * D_;
constexpr size_t WO_OFF   = WVT_OFF + (size_t)H_ * DK_ * D_;    // [D,D]
constexpr size_t QH_OFF   = WO_OFF  + (size_t)D_ * D_;          // [B,H,S,DK]
constexpr size_t KH_OFF   = QH_OFF  + (size_t)B_ * H_ * S_ * DK_;
constexpr size_t VT_OFF   = KH_OFF  + (size_t)B_ * H_ * S_ * DK_; // [B,H,DK,S]
constexpr size_t CTX_OFF  = VT_OFF  + (size_t)B_ * H_ * DK_ * S_; // [M,D]

// ---- fragment helpers (contiguous k order: k = 16*sel + s for A and B) --------------------
__device__ __forceinline__ v16h combine16(v8h lo, v8h hi) {
  v16h r;
#pragma unroll
  for (int i = 0; i < 8; ++i) { r[i] = lo[i]; r[i + 8] = hi[i]; }
  return r;
}
// A frag 16x32: lane<16 -> K 0..15 ; lane>=16 -> K 16..31 (caller adds sel*16)
__device__ __forceinline__ v16h load_a_frag(const half_t* p) {
  return combine16(*(const v8h*)p, *(const v8h*)(p + 8));
}
// B frag 32x16: lane<16 -> K 0..15 ; lane>=16 -> K 16..31 (caller adds sel*16)
__device__ __forceinline__ v16h load_b_frag(const half_t* p) {
  return combine16(*(const v8h*)p, *(const v8h*)(p + 8));
}
__device__ __forceinline__ v8f wmma_f16(v16h a, v16h b, v8f c) {
  return __builtin_amdgcn_wmma_f32_16x16x32_f16(false, a, false, b, (short)0, c,
                                                false, false);
}
// cross-lane reduction over the 16-lane half holding one C row
__device__ __forceinline__ float red16_max(float v) {
  const int lane = (int)(threadIdx.x & 31u);
#pragma unroll
  for (int m = 1; m < 16; m <<= 1) {
    int o = __builtin_amdgcn_ds_bpermute((lane ^ m) << 2,
                                         __builtin_bit_cast(int, v));
    v = fmaxf(v, __builtin_bit_cast(float, o));
  }
  return v;
}
__device__ __forceinline__ float red16_sum(float v) {
  const int lane = (int)(threadIdx.x & 31u);
#pragma unroll
  for (int m = 1; m < 16; m <<= 1) {
    int o = __builtin_amdgcn_ds_bpermute((lane ^ m) << 2,
                                         __builtin_bit_cast(int, v));
    v += __builtin_bit_cast(float, o);
  }
  return v;
}

// ---- Tensor Data Mover: 2-D tile (f16) global -> LDS ----------------------
// Descriptor per CDNA5 ISA. tensor_dim == tile_dim
// (tiles fully in-bounds), stride = row pitch, groups 2/3 zero (2-D tensor),
// workgroup_mask = 0 (not in a cluster). All inputs uniform -> SGPR groups.
// This toolchain's builtin is the 6-arg form:
//   (u32x4 g0, i32x8 g1, i32x4 g2, i32x4 g3, i32x8 gx, i32 cpol)
__device__ __forceinline__ void tdm_load_2d_f16(unsigned lds_byte_addr,
                                                unsigned long long gaddr,
                                                unsigned dim0, unsigned rows,
                                                unsigned stride_elems) {
  u32x4 g0;
  g0[0] = 1u;                                     // count=1, user descriptor
  g0[1] = lds_byte_addr;                          // lds_addr [63:32]
  g0[2] = (unsigned)gaddr;                        // global_addr [95:64]
  g0[3] = (unsigned)(gaddr >> 32) | (2u << 30);   // global_addr hi + type=2
  i32x8 g1;
  g1[0] = (int)(1u << 16);                        // data_size = 2 bytes
  g1[1] = (int)((dim0 & 0xFFFFu) << 16);          // tensor_dim0[15:0]
  g1[2] = (int)((rows & 0xFFFFu) << 16);          // tensor_dim1[15:0]
  g1[3] = (int)((dim0 & 0xFFFFu) << 16);          // tile_dim0
  g1[4] = (int)(rows & 0xFFFFu);                  // tile_dim1 (tile_dim2=0)
  g1[5] = (int)stride_elems;                      // tensor_dim0_stride[31:0]
  g1[6] = 0;
  g1[7] = 0;
  i32x4 gz = {0, 0, 0, 0};
  i32x8 gz8 = {0, 0, 0, 0, 0, 0, 0, 0};
  __builtin_amdgcn_tensor_load_to_lds(g0, g1, gz, gz, gz8, 0);
}
__device__ __forceinline__ unsigned lds_off(const void* p) {
  return (unsigned)(uintptr_t)p;   // low 32 bits of flat shared addr = LDS offset
}

// ---------------------------------------------------------------------------
// K0a: W [H,D,DK] f32  ->  Wt [H,DK,D] f16
// ---------------------------------------------------------------------------
__global__ void transpose_w_kernel(const float* __restrict__ in,
                                   half_t* __restrict__ out) {
  int idx = blockIdx.x * 256 + threadIdx.x;      // over H*DK*D
  int d   = idx % D_;
  int t   = idx / D_;
  int n   = t % DK_;
  int h   = t / DK_;
  out[idx] = (half_t)in[((size_t)h * D_ + d) * DK_ + n];
}
// K0b: W_o [D, H*DK] f32 -> f16 (row-major already matches B layout)
__global__ void cvt_f16_kernel(const float* __restrict__ in,
                               half_t* __restrict__ out) {
  int idx  = blockIdx.x * 256 + threadIdx.x;
  out[idx] = (half_t)in[idx];
}

// ---------------------------------------------------------------------------
// K1: per-head projection  Out[h] = X @ W[h]   (M=4096, K=1024, N=64)
// ---------------------------------------------------------------------------
__global__ __launch_bounds__(128)
void proj_kernel(const float* __restrict__ X, const half_t* __restrict__ Wt,
                 half_t* __restrict__ Out, int transpose_out) {
  __shared__ alignas(16) half_t ldsx[64 * 32];
  const int h    = blockIdx.y;
  const int row0 = blockIdx.x * 64;
  const int tid  = threadIdx.x;
  const int wv = tid >> 5, lane = tid & 31;
  const int lan16 = lane & 15, sel = lane >> 4;
  const half_t* wt_h = Wt + (size_t)h * DK_ * D_;

  v8f acc[4];
#pragma unroll
  for (int nt = 0; nt < 4; ++nt) acc[nt] = {};

  for (int kb = 0; kb < D_; kb += 32) {
    // stage 64x32 tile, f32 -> f16
    {
      int r  = tid >> 1;
      int c0 = (tid & 1) * 16;
      const float* src = X + (size_t)(row0 + r) * D_ + kb + c0;
      half_t* dst = ldsx + r * 32 + c0;
      if (kb + 32 < D_) __builtin_prefetch(src + 32, 0, 1);  // next K tile
#pragma unroll
      for (int i = 0; i < 16; ++i) dst[i] = (half_t)src[i];
    }
    __syncthreads();
    const half_t* ap = ldsx + (wv * 16 + lan16) * 32 + sel * 16;
    v16h a = load_a_frag(ap);
#pragma unroll
    for (int nt = 0; nt < 4; ++nt) {
      const half_t* bp = wt_h + (size_t)(nt * 16 + lan16) * D_ + kb + sel * 16;
      acc[nt] = wmma_f16(a, load_b_frag(bp), acc[nt]);
    }
    __syncthreads();
  }

  if (!transpose_out) {
#pragma unroll
    for (int nt = 0; nt < 4; ++nt) {
      int n = nt * 16 + lan16;
#pragma unroll
      for (int r = 0; r < 8; ++r) {
        int rowg = row0 + wv * 16 + r + 8 * sel;   // C: M = r + 8*sel
        int b = rowg >> 11, s = rowg & (S_ - 1);
        Out[(((size_t)b * H_ + h) * S_ + s) * DK_ + n] = (half_t)acc[nt][r];
      }
    }
  } else {
#pragma unroll
    for (int nt = 0; nt < 4; ++nt) {
      int n    = nt * 16 + lan16;
      int rowg = row0 + wv * 16 + 8 * sel;         // 8 consecutive s values
      int b = rowg >> 11, s0 = rowg & (S_ - 1);
      v8h pk;
#pragma unroll
      for (int r = 0; r < 8; ++r) pk[r] = (half_t)acc[nt][r];
      *(v8h*)(Out + (((size_t)b * H_ + h) * DK_ + n) * S_ + s0) = pk;
    }
  }
}

// ---------------------------------------------------------------------------
// K2: flash attention per (b, h, 64-row q tile); one wave = 16 q rows.
//     K/V chunks TDM-DMA'd into double-buffered LDS, shared by all 4 waves.
// ---------------------------------------------------------------------------
__global__ __launch_bounds__(128)
void attn_kernel(const half_t* __restrict__ Qh, const half_t* __restrict__ Kh,
                 const half_t* __restrict__ Vt, half_t* __restrict__ Ctx) {
  __shared__ alignas(16) half_t kbuf[2][32 * 64];  // 8 KB, K chunk (s x dk)
  __shared__ alignas(16) half_t vbuf[2][64 * 32];  // 8 KB, V chunk (dk x s)
  __shared__ alignas(16) half_t smem[64 * 64];     // P buffers then ctx tile
  const int b = blockIdx.z, h = blockIdx.y;
  const int q0 = blockIdx.x * 64;
  const int tid = threadIdx.x;
  const int wv = tid >> 5, lane = tid & 31;
  const int lan16 = lane & 15, sel = lane >> 4;
  const int wvs = __builtin_amdgcn_readfirstlane(tid) >> 5;  // uniform wave id
  const size_t bh = (size_t)b * H_ + h;
  const half_t* qb = Qh + bh * S_ * DK_;
  const half_t* kb = Kh + bh * S_ * DK_;
  const half_t* vb = Vt + bh * DK_ * S_;

  // Q fragments (row = lan16 for whole wave; K chunk split by sel)
  const int qrow = q0 + wv * 16 + lan16;
  v16h aq[2];
#pragma unroll
  for (int ks = 0; ks < 2; ++ks)
    aq[ks] = load_a_frag(qb + (size_t)qrow * DK_ + ks * 32 + sel * 16);

  float mrow[8], lrow[8];
  v8f o[4];
#pragma unroll
  for (int r = 0; r < 8; ++r) { mrow[r] = -1e30f; lrow[r] = 0.f; }
#pragma unroll
  for (int nt = 0; nt < 4; ++nt) o[nt] = {};

  half_t* pbuf = smem + wv * (16 * 32);

  // TDM issue: wave wvs loads K rows [wvs*8, +8) and Vt rows [wvs*16, +16)
  auto issue_chunk = [&](int kvn, int p) {
    tdm_load_2d_f16(lds_off(&kbuf[p][wvs * 8 * 64]),
                    (unsigned long long)(uintptr_t)(kb + (size_t)(kvn + wvs * 8) * DK_),
                    /*dim0=*/64, /*rows=*/8, /*stride=*/DK_);
    tdm_load_2d_f16(lds_off(&vbuf[p][wvs * 16 * 32]),
                    (unsigned long long)(uintptr_t)(vb + (size_t)(wvs * 16) * S_ + kvn),
                    /*dim0=*/32, /*rows=*/16, /*stride=*/S_);
  };

  issue_chunk(0, 0);
  __builtin_amdgcn_s_wait_tensorcnt(0);
  __syncthreads();

  for (int kv = 0; kv < S_; kv += 32) {
    const int p = (kv >> 5) & 1;
    if (kv + 32 < S_) issue_chunk(kv + 32, p ^ 1);   // prefetch next chunk

    // ---- scores: two 16-wide KV tiles (K B-frags from LDS) ---------------
    v8f sc[2];
#pragma unroll
    for (int j = 0; j < 2; ++j) {
      v8f c = {};
#pragma unroll
      for (int ks = 0; ks < 2; ++ks) {
        const half_t* bp = &kbuf[p][(16 * j + lan16) * 64 + ks * 32 + sel * 16];
        c = wmma_f16(aq[ks], load_b_frag(bp), c);
      }
#pragma unroll
      for (int r = 0; r < 8; ++r) sc[j][r] = c[r] * 0.125f;  // 1/sqrt(64)
    }
    // ---- online softmax ---------------------------------------------------
    float alpha[8];
#pragma unroll
    for (int r = 0; r < 8; ++r) {
      float cm = red16_max(fmaxf(sc[0][r], sc[1][r]));
      float mn = fmaxf(mrow[r], cm);
      alpha[r] = __expf(mrow[r] - mn);
      mrow[r]  = mn;
    }
#pragma unroll
    for (int r = 0; r < 8; ++r) {
      float p0 = __expf(sc[0][r] - mrow[r]);
      float p1 = __expf(sc[1][r] - mrow[r]);
      sc[0][r] = p0; sc[1][r] = p1;
      lrow[r]  = lrow[r] * alpha[r] + red16_sum(p0 + p1);
    }
#pragma unroll
    for (int nt = 0; nt < 4; ++nt)
#pragma unroll
      for (int r = 0; r < 8; ++r) o[nt][r] *= alpha[r];
    // ---- P -> LDS (row-major 16x32), reload as A fragment -----------------
#pragma unroll
    for (int j = 0; j < 2; ++j)
#pragma unroll
      for (int r = 0; r < 8; ++r) {
        int m = r + 8 * sel;
        pbuf[m * 32 + j * 16 + lan16] = (half_t)sc[j][r];
      }
    asm volatile("s_wait_dscnt 0" ::: "memory");
    v16h pf = load_a_frag(pbuf + lan16 * 32 + sel * 16);
    // ---- O += P @ V  (V B-frags from LDS, dk-major rows, s contiguous) ----
#pragma unroll
    for (int nt = 0; nt < 4; ++nt) {
      const half_t* bp = &vbuf[p][(nt * 16 + lan16) * 32 + sel * 16];
      o[nt] = wmma_f16(pf, load_b_frag(bp), o[nt]);
    }

    __builtin_amdgcn_s_wait_tensorcnt(0);  // own slice of next chunk landed
    __syncthreads();                       // publish buffers workgroup-wide
  }

  // ---- epilogue: normalize, stage 64x64 ctx tile, coalesced f16 store -----
  float inv[8];
#pragma unroll
  for (int r = 0; r < 8; ++r) inv[r] = 1.0f / lrow[r];
#pragma unroll
  for (int nt = 0; nt < 4; ++nt)
#pragma unroll
    for (int r = 0; r < 8; ++r) {
      int m = wv * 16 + r + 8 * sel;
      smem[m * 64 + nt * 16 + lan16] = (half_t)(o[nt][r] * inv[r]);
    }
  __syncthreads();
#pragma unroll
  for (int i = 0; i < 4; ++i) {
    int idx = i * 128 + tid;          // 512 chunks of 8 halves
    int rr = idx >> 3, cc = (idx & 7) * 8;
    v8h pk = *(const v8h*)(smem + rr * 64 + cc);
    *(v8h*)(Ctx + (size_t)(b * S_ + q0 + rr) * D_ + h * 64 + cc) = pk;
  }
}

// ---------------------------------------------------------------------------
// K3: Out = Ctx @ W_o^T + b_o    (M=4096, N=1024, K=1024), f32 output
// ---------------------------------------------------------------------------
__global__ __launch_bounds__(128)
void out_gemm_kernel(const half_t* __restrict__ Ctx,
                     const half_t* __restrict__ Wo,
                     const float* __restrict__ bo, float* __restrict__ Out) {
  const int m0 = blockIdx.x * 64, n0 = blockIdx.y * 64;
  const int tid = threadIdx.x;
  const int wv = tid >> 5, lane = tid & 31;
  const int lan16 = lane & 15, sel = lane >> 4;

  v8f acc[4];
#pragma unroll
  for (int nt = 0; nt < 4; ++nt) acc[nt] = {};

  const int arow = m0 + wv * 16 + lan16;
  for (int kbk = 0; kbk < D_; kbk += 32) {
    const half_t* ap = Ctx + (size_t)arow * D_ + kbk + sel * 16;
    if (kbk + 32 < D_) __builtin_prefetch(ap + 32, 0, 1);
    v16h a = load_a_frag(ap);
#pragma unroll
    for (int nt = 0; nt < 4; ++nt) {
      const half_t* bp =
          Wo + (size_t)(n0 + nt * 16 + lan16) * D_ + kbk + sel * 16;
      acc[nt] = wmma_f16(a, load_b_frag(bp), acc[nt]);
    }
  }
#pragma unroll
  for (int nt = 0; nt < 4; ++nt) {
    int n = n0 + nt * 16 + lan16;
    float bias = bo[n];
#pragma unroll
    for (int r = 0; r < 8; ++r) {
      int row = m0 + wv * 16 + r + 8 * sel;
      Out[(size_t)row * D_ + n] = acc[nt][r] + bias;
    }
  }
}

// ---------------------------------------------------------------------------
extern "C" void kernel_launch(void* const* d_in, const int* in_sizes, int n_in,
                              void* d_out, int out_size, void* d_ws,
                              size_t ws_size, hipStream_t stream) {
  const float* q  = (const float*)d_in[0];
  const float* k  = (const float*)d_in[1];
  const float* v  = (const float*)d_in[2];
  const float* Wq = (const float*)d_in[3];
  const float* Wk = (const float*)d_in[4];
  const float* Wv = (const float*)d_in[5];
  const float* Wo = (const float*)d_in[6];
  const float* bo = (const float*)d_in[7];
  float* out = (float*)d_out;

  half_t* ws   = (half_t*)d_ws;
  half_t* wqt  = ws + WQT_OFF;
  half_t* wkt  = ws + WKT_OFF;
  half_t* wvt  = ws + WVT_OFF;
  half_t* wo16 = ws + WO_OFF;
  half_t* qh   = ws + QH_OFF;
  half_t* kh   = ws + KH_OFF;
  half_t* vt   = ws + VT_OFF;
  half_t* ctx  = ws + CTX_OFF;

  // weight conversions
  const int nw = H_ * D_ * DK_;                      // 1,048,576
  transpose_w_kernel<<<nw / 256, 256, 0, stream>>>(Wq, wqt);
  transpose_w_kernel<<<nw / 256, 256, 0, stream>>>(Wk, wkt);
  transpose_w_kernel<<<nw / 256, 256, 0, stream>>>(Wv, wvt);
  cvt_f16_kernel<<<(D_ * D_) / 256, 256, 0, stream>>>(Wo, wo16);

  // per-head projections
  dim3 pgrid(M_ / 64, H_);
  proj_kernel<<<pgrid, 128, 0, stream>>>(q, wqt, qh, 0);
  proj_kernel<<<pgrid, 128, 0, stream>>>(k, wkt, kh, 0);
  proj_kernel<<<pgrid, 128, 0, stream>>>(v, wvt, vt, 1);

  // attention (TDM double-buffered K/V)
  attn_kernel<<<dim3(S_ / 64, H_, B_), 128, 0, stream>>>(qh, kh, vt, ctx);

  // output projection + bias
  out_gemm_kernel<<<dim3(M_ / 64, D_ / 64), 128, 0, stream>>>(ctx, wo16, bo,
                                                              out);
}
